// ReasoningLayer_81939386073751
// MI455X (gfx1250) — compile-verified, matrix operand order checked by reading, not run on hardware
//
#include <hip/hip_runtime.h>
#include <math.h>

// ---------------------------------------------------------------------------
// MoE "ReasoningLayer" for MI455X (gfx1250, wave32, WMMA, TDM).
//   out = x + sum_e softmax(x@Wr)[e] * ( gelu(x@Wu[e]+bu[e]) @ Wd[e] + bd[e] )
// bf16 WMMA (v_wmma_f32_16x16x32_bf16) both GEMM stages, fp32 accumulation.
// K-loop data movement via Tensor Data Mover (tensor_load_to_lds) with
// double-buffered LDS; TDM hardware padding reproduces the 80B LDS pitch.
// ---------------------------------------------------------------------------

typedef __attribute__((ext_vector_type(16))) __bf16 v16bf;
typedef __attribute__((ext_vector_type(8)))  __bf16 v8bf;
typedef __attribute__((ext_vector_type(8)))  float  v8f;
typedef __attribute__((ext_vector_type(4)))  unsigned int u32x4;
typedef __attribute__((ext_vector_type(8)))  int i32x8;
typedef __attribute__((ext_vector_type(4)))  int i32x4;

#define DIM    2048
#define NEXP   8
#define EDIM   1024
#define TOKENS 8192   // B * S
#define LDS_PITCH 40  // bf16 elems per row: 64B data + 16B pad = 80B

#if __has_builtin(__builtin_amdgcn_tensor_load_to_lds) && \
    __has_builtin(__builtin_amdgcn_s_wait_tensorcnt)
#define USE_TDM 1
#else
#define USE_TDM 0
#endif

// ----------------------------- helpers -------------------------------------

__device__ __forceinline__ float gelu_tanh(float v) {
  // 0.5*v*(1+tanh(c)) == v * sigmoid(2c); branch-free (v_exp_f32 + v_rcp_f32)
  float c = 0.7978845608028654f * (v + 0.044715f * v * v * v);
  float e = __expf(-2.0f * c);
  return v * __builtin_amdgcn_rcpf(1.0f + e);
}

#if USE_TDM
// Issue one TDM 2D-tile load: 32 bf16 (64B) x 128 rows from global (row stride
// = rowStride elems) into LDS at lds_off, padding +16B after every 64B row so
// the LDS image has an 80B pitch (LDS_PITCH). D# packing per CDNA5 ISA 8.3/8.4.
__device__ __forceinline__ void tdm_load_tile(const __bf16* gptr, unsigned lds_off,
                                              unsigned rowStride) {
  unsigned long long ga = (unsigned long long)(const void*)gptr;
  u32x4 g0;
  g0[0] = 1u;                                            // count=1 (valid), user D#
  g0[1] = lds_off;                                       // lds_addr (bytes)
  g0[2] = (unsigned)(ga & 0xFFFFFFFFu);                  // global_addr[31:0]
  g0[3] = (unsigned)((ga >> 32) & 0x01FFFFFFu)           // global_addr[56:32]
          | (2u << 30);                                  // type=2 ("image")
  i32x8 g1;
  unsigned long long s0 = (unsigned long long)rowStride; // tensor_dim0_stride
  g1[0] = (int)((1u << 16)      // data_size = 1 -> 2 bytes
              | (1u << 20)      // pad_enable
              | (3u << 22)      // pad_interval = 16 DWORDs (64B)
              | (3u << 25));    // pad_amount   = 4 DWORDs (16B)
  g1[1] = (int)((rowStride & 0xFFFFu) << 16);            // tensor_dim0[15:0]
  g1[2] = (int)(((rowStride >> 16) & 0xFFFFu)            // tensor_dim0[31:16]
              | (0xFFFFu << 16));                        // tensor_dim1[15:0]
  g1[3] = (int)(0x7FFFu | (32u << 16));                  // dim1[31:16] big; tile_dim0=32
  g1[4] = (int)(128u);                                   // tile_dim1=128, tile_dim2=0
  g1[5] = (int)(unsigned)(s0 & 0xFFFFFFFFu);             // dim0_stride[31:0]
  g1[6] = (int)(unsigned)((s0 >> 32) & 0xFFFFu);         // dim0_stride[47:32]
  g1[7] = 0;
  i32x4 z4 = {0, 0, 0, 0};
  i32x8 z8 = {0, 0, 0, 0, 0, 0, 0, 0};
  // 6-arg form (clang-23 / therock-10.0 headers): groups 2/3 + extra group zero
  __builtin_amdgcn_tensor_load_to_lds(g0, g1, z4, z4, z8, 0);
}
__device__ __forceinline__ unsigned lds_offset_of(const void* p) {
  // flat LDS-aperture address: low 32 bits are the LDS byte offset
  return (unsigned)(unsigned long long)p;
}
#endif

// ------------------------- elementwise kernels ------------------------------

__global__ void cvt_bf16_kernel(const float* __restrict__ in,
                                __bf16* __restrict__ out, long n) {
  long i = ((long)blockIdx.x * blockDim.x + threadIdx.x) * 4;
  if (i + 3 < n) {
    float4 v = *(const float4*)(in + i);
    __bf16 t[4];
    t[0] = (__bf16)v.x; t[1] = (__bf16)v.y; t[2] = (__bf16)v.z; t[3] = (__bf16)v.w;
    *(uint2*)(out + i) = *(const uint2*)t;
  }
}

// fp32 [R,C] -> bf16 [C,R] (per-expert slice via blockIdx.z), LDS-tiled.
__global__ __launch_bounds__(256)
void transpose_cvt_kernel(const float* __restrict__ in,
                          __bf16* __restrict__ out, int R, int C) {
  __shared__ __bf16 tile[32][33];
  const long slice = (long)R * C;
  in  += (long)blockIdx.z * slice;
  out += (long)blockIdx.z * slice;
  const int r0 = blockIdx.y * 32;
  const int c0 = blockIdx.x * 32;
  const int tx = threadIdx.x & 31;
  const int ty = threadIdx.x >> 5;
#pragma unroll
  for (int i = 0; i < 4; ++i) {
    int r = r0 + ty + i * 8;
    tile[ty + i * 8][tx] = (__bf16)in[(long)r * C + c0 + tx];
  }
  __syncthreads();
#pragma unroll
  for (int i = 0; i < 4; ++i) {
    int c = c0 + ty + i * 8;
    out[(long)c * R + r0 + tx] = tile[tx][ty + i * 8];
  }
}

__global__ void copy_f32_kernel(const float* __restrict__ in,
                                float* __restrict__ out, long n) {
  long i = ((long)blockIdx.x * blockDim.x + threadIdx.x) * 4;
  if (i + 3 < n) *(float4*)(out + i) = *(const float4*)(in + i);
}

// ----------------------------- router ---------------------------------------

__global__ void router_kernel(const float* __restrict__ x,
                              const float* __restrict__ Wr,
                              const float* __restrict__ br,
                              float* __restrict__ probs) {
  int t = blockIdx.x * blockDim.x + threadIdx.x;
  if (t >= TOKENS) return;
  float acc[NEXP];
#pragma unroll
  for (int e = 0; e < NEXP; ++e) acc[e] = br[e];
  const float* xr = x + (long)t * DIM;
  for (int d = 0; d < DIM; ++d) {
    float xv = xr[d];
#pragma unroll
    for (int e = 0; e < NEXP; ++e) acc[e] = fmaf(xv, Wr[d * NEXP + e], acc[e]);
  }
  float m = acc[0];
#pragma unroll
  for (int e = 1; e < NEXP; ++e) m = fmaxf(m, acc[e]);
  float s = 0.0f;
#pragma unroll
  for (int e = 0; e < NEXP; ++e) { acc[e] = __expf(acc[e] - m); s += acc[e]; }
  float inv = 1.0f / s;
#pragma unroll
  for (int e = 0; e < NEXP; ++e) probs[(long)t * NEXP + e] = acc[e] * inv;
}

// ----------------------- WMMA GEMM (bf16 -> f32) -----------------------------
// Block tile 128x128, 256 threads = 8 wave32s (2 M x 4 N); wave tile 64x32 =
// 4x2 wmma 16x16 tiles, K-step 32. Fragment layout per ISA 7.12.2: two 16B
// LDS chunks per lane at k-offsets {hi8*8, hi8*8+16}, 80B row pitch.

__device__ __forceinline__ v16bf load_frag(const __bf16* lds, int row, int hi8) {
  v16bf f;
  const __bf16* p = lds + row * LDS_PITCH + hi8 * 8;
  ((v8bf*)&f)[0] = *(const v8bf*)(p);
  ((v8bf*)&f)[1] = *(const v8bf*)(p + 16);
  return f;
}

// Shared mainloop body: accumulate A[128,KD] x B^T[128,KD] block tile.
// Apan/Bpan point at the block's first row (row stride = KD elems).
template <int KD>
__device__ __forceinline__ void gemm_mainloop(const __bf16* __restrict__ Apan,
                                              const __bf16* __restrict__ Bpan,
                                              __bf16 (&lA)[2][128 * LDS_PITCH],
                                              __bf16 (&lB)[2][128 * LDS_PITCH],
                                              int tid, int wave, int lane,
                                              int wm, int wn, v8f (&acc)[4][2]) {
  const int hi8 = (lane >> 4) & 1;
  const int l15 = lane & 15;
#if USE_TDM
  if (wave == 0) {   // TDM issue is wave-level (EXEC ignored); one wave drives it
    tdm_load_tile(Apan, lds_offset_of(&lA[0][0]), KD);
    tdm_load_tile(Bpan, lds_offset_of(&lB[0][0]), KD);
  }
  int cur = 0;
  for (int k0 = 0; k0 < KD; k0 += 32) {
    if (wave == 0) {
      if (k0 + 32 < KD) {   // stream next tile into the other buffer
        tdm_load_tile(Apan + k0 + 32, lds_offset_of(&lA[cur ^ 1][0]), KD);
        tdm_load_tile(Bpan + k0 + 32, lds_offset_of(&lB[cur ^ 1][0]), KD);
        __builtin_amdgcn_s_wait_tensorcnt(2);  // current tile's 2 DMAs retired
      } else {
        __builtin_amdgcn_s_wait_tensorcnt(0);
      }
    }
    __syncthreads();       // publish current tile to all waves
    v16bf a[4], b[2];
#pragma unroll
    for (int i = 0; i < 4; ++i) a[i] = load_frag(&lA[cur][0], wm * 64 + i * 16 + l15, hi8);
#pragma unroll
    for (int j = 0; j < 2; ++j) b[j] = load_frag(&lB[cur][0], wn * 32 + j * 16 + l15, hi8);
#pragma unroll
    for (int i = 0; i < 4; ++i)
#pragma unroll
      for (int j = 0; j < 2; ++j)
        acc[i][j] = __builtin_amdgcn_wmma_f32_16x16x32_bf16(
            false, a[i], false, b[j], (short)0, acc[i][j], false, false);
    __syncthreads();       // all reads done before buffer is overwritten
    cur ^= 1;
  }
#else
  for (int k0 = 0; k0 < KD; k0 += 32) {
#pragma unroll
    for (int i = 0; i < 2; ++i) {
      int id = tid + i * 256;
      int row = id >> 2, c = id & 3;
      *(uint4*)&lA[0][row * LDS_PITCH + c * 8] =
          *(const uint4*)(Apan + (long)row * KD + k0 + c * 8);
      *(uint4*)&lB[0][row * LDS_PITCH + c * 8] =
          *(const uint4*)(Bpan + (long)row * KD + k0 + c * 8);
    }
    __syncthreads();
    v16bf a[4], b[2];
#pragma unroll
    for (int i = 0; i < 4; ++i) a[i] = load_frag(&lA[0][0], wm * 64 + i * 16 + l15, hi8);
#pragma unroll
    for (int j = 0; j < 2; ++j) b[j] = load_frag(&lB[0][0], wn * 32 + j * 16 + l15, hi8);
#pragma unroll
    for (int i = 0; i < 4; ++i)
#pragma unroll
      for (int j = 0; j < 2; ++j)
        acc[i][j] = __builtin_amdgcn_wmma_f32_16x16x32_bf16(
            false, a[i], false, b[j], (short)0, acc[i][j], false, false);
    __syncthreads();
  }
#endif
}

// ------------------------------- up GEMM -------------------------------------
// H[t, n] = gelu( X[t,:] . WuT[n,:] + bu[n] )   (K = DIM)
__global__ __launch_bounds__(256)
void moe_up_kernel(const __bf16* __restrict__ X,     // [TOKENS, DIM]
                   const __bf16* __restrict__ WuT,   // [EDIM, DIM] (this expert)
                   const float*  __restrict__ bu,    // [EDIM]      (this expert)
                   __bf16* __restrict__ H) {         // [TOKENS, EDIM]
  __shared__ __align__(16) __bf16 lA[2][128 * LDS_PITCH];
  __shared__ __align__(16) __bf16 lB[2][128 * LDS_PITCH];
  const int tid = threadIdx.x, lane = tid & 31, wave = tid >> 5;
  const int wm = wave & 1, wn = wave >> 1;
  const int mBase = blockIdx.y * 128, nBase = blockIdx.x * 128;
  const int hi8 = (lane >> 4) & 1, l15 = lane & 15;

  v8f acc[4][2];
#pragma unroll
  for (int i = 0; i < 4; ++i)
#pragma unroll
    for (int j = 0; j < 2; ++j) acc[i][j] = (v8f)(0.0f);

  gemm_mainloop<DIM>(X + (long)mBase * DIM, WuT + (long)nBase * DIM,
                     lA, lB, tid, wave, lane, wm, wn, acc);

#pragma unroll
  for (int j = 0; j < 2; ++j) {
    int n = nBase + wn * 32 + j * 16 + l15;
    float buv = bu[n];
#pragma unroll
    for (int i = 0; i < 4; ++i) {
      int rowT = mBase + wm * 64 + i * 16 + hi8 * 8;
#pragma unroll
      for (int r = 0; r < 8; ++r) {
        float v = acc[i][j][r] + buv;
        H[(long)(rowT + r) * EDIM + n] = (__bf16)gelu_tanh(v);
      }
    }
  }
}

// ------------------------------ down GEMM ------------------------------------
// Out[t, n] += probs[t,e] * ( H[t,:] . WdT[n,:] + bd[n] )   (K = EDIM)
__global__ __launch_bounds__(256)
void moe_down_kernel(const __bf16* __restrict__ H,     // [TOKENS, EDIM]
                     const __bf16* __restrict__ WdT,   // [DIM, EDIM] (this expert)
                     const float*  __restrict__ bd,    // [DIM]       (this expert)
                     const float*  __restrict__ probs, // [TOKENS, NEXP]
                     int expert,
                     float* __restrict__ Out) {        // [TOKENS, DIM]
  __shared__ __align__(16) __bf16 lA[2][128 * LDS_PITCH];
  __shared__ __align__(16) __bf16 lB[2][128 * LDS_PITCH];
  const int tid = threadIdx.x, lane = tid & 31, wave = tid >> 5;
  const int wm = wave & 1, wn = wave >> 1;
  const int mBase = blockIdx.y * 128, nBase = blockIdx.x * 128;
  const int hi8 = (lane >> 4) & 1, l15 = lane & 15;

  v8f acc[4][2];
#pragma unroll
  for (int i = 0; i < 4; ++i)
#pragma unroll
    for (int j = 0; j < 2; ++j) acc[i][j] = (v8f)(0.0f);

  gemm_mainloop<EDIM>(H + (long)mBase * EDIM, WdT + (long)nBase * EDIM,
                      lA, lB, tid, wave, lane, wm, wn, acc);

  // Out += p * (acc + bd). Blocks own disjoint tiles; expert kernels are
  // stream-ordered, so plain RMW is race-free and deterministic.
#pragma unroll
  for (int j = 0; j < 2; ++j) {
    int n = nBase + wn * 32 + j * 16 + l15;
    float bdv = bd[n];
#pragma unroll
    for (int i = 0; i < 4; ++i) {
      int rowT = mBase + wm * 64 + i * 16 + hi8 * 8;
#pragma unroll
      for (int r = 0; r < 8; ++r) {
        int row = rowT + r;
        float p = probs[(long)row * NEXP + expert];
        Out[(long)row * DIM + n] += p * (acc[i][j][r] + bdv);
      }
    }
  }
}

// ------------------------------- launcher ------------------------------------

extern "C" void kernel_launch(void* const* d_in, const int* in_sizes, int n_in,
                              void* d_out, int out_size, void* d_ws, size_t ws_size,
                              hipStream_t stream) {
  const float* x  = (const float*)d_in[0];   // [4,2048,2048]
  const float* Wr = (const float*)d_in[1];   // [2048,8]
  const float* br = (const float*)d_in[2];   // [8]
  const float* Wu = (const float*)d_in[3];   // [8,2048,1024]
  const float* bu = (const float*)d_in[4];   // [8,1024]
  const float* Wd = (const float*)d_in[5];   // [8,1024,2048]
  const float* bd = (const float*)d_in[6];   // [8,2048]
  float* out = (float*)d_out;                // [4,2048,2048]

  char* ws = (char*)d_ws;
  __bf16* xb    = (__bf16*)(ws);                 // 32 MB
  __bf16* WuT   = (__bf16*)(ws + (32l << 20));   // 32 MB  [e][1024][2048]
  __bf16* WdT   = (__bf16*)(ws + (64l << 20));   // 32 MB  [e][2048][1024]
  float*  probs = (float*) (ws + (96l << 20));   // 256 KB [8192][8]
  __bf16* Hbuf  = (__bf16*)(ws + (97l << 20));   // 16 MB  [8192][1024]

  const long nX = (long)TOKENS * DIM;

  cvt_bf16_kernel<<<dim3((unsigned)(nX / 4 / 256)), dim3(256), 0, stream>>>(x, xb, nX);
  transpose_cvt_kernel<<<dim3(EDIM / 32, DIM / 32, NEXP), dim3(256), 0, stream>>>(
      Wu, WuT, DIM, EDIM);
  transpose_cvt_kernel<<<dim3(DIM / 32, EDIM / 32, NEXP), dim3(256), 0, stream>>>(
      Wd, WdT, EDIM, DIM);
  router_kernel<<<dim3(TOKENS / 256), dim3(256), 0, stream>>>(x, Wr, br, probs);
  copy_f32_kernel<<<dim3((unsigned)(nX / 4 / 256)), dim3(256), 0, stream>>>(x, out, nX);

  for (int e = 0; e < NEXP; ++e) {
    moe_up_kernel<<<dim3(EDIM / 128, TOKENS / 128), dim3(256), 0, stream>>>(
        xb, WuT + (long)e * EDIM * DIM, bu + (long)e * EDIM, Hbuf);
    moe_down_kernel<<<dim3(DIM / 128, TOKENS / 128), dim3(256), 0, stream>>>(
        Hbuf, WdT + (long)e * DIM * EDIM, bd + (long)e * DIM, probs, e, out);
  }
}
